// GraphAttentionEmbedding_7404523618674
// MI455X (gfx1250) — matrix operand order repaired and checked by reading, not compiled, not run on hardware
//
#include <hip/hip_runtime.h>
#include <hip/hip_bf16.h>
#include <math.h>

#define N_NODES 50000
#define N_EDGES 800000
#define IN_DIM  128
#define EDGE_DIM 32
#define C1      128   // heads(4) * hid(32)
#define OUT_DIM 64

typedef __attribute__((ext_vector_type(16))) _Float16 v16h;
typedef __attribute__((ext_vector_type(8)))  float    v8f;

// ---------------------------------------------------------------------------
// WMMA GEMM: C[M x (gridDim.x*16)] = A[M x K] @ W[K x *] (+ bias | += )
// fp32 io, f16 WMMA inner, K templated (32/64/128) -> fully unrolled wmma.
// blockIdx.z selects a "head": A/W/C pointers advance by per-head strides.
// Weight tile is staged in LDS in B-FRAGMENT ORDER so each lane loads its
// v16h fragment as two contiguous ds_load_b128.
// ---------------------------------------------------------------------------
template <int K, bool ACC>
__global__ __launch_bounds__(128) void gemm_wmma(
    const float* __restrict__ A, int lda, int aHead,
    const float* __restrict__ W, int ldw, int wHead,
    const float* __restrict__ bias,
    float* __restrict__ C, int ldc, int cHead, int M)
{
  __shared__ __align__(32) _Float16 lw[2048];   // K<=128: K*16 halves
  const int n0 = blockIdx.x * 16;
  A += (size_t)blockIdx.z * aHead;
  W += (size_t)blockIdx.z * wHead;
  C += (size_t)blockIdx.z * cHead;

  // Stage W column tile in fragment order:
  //   lw[chunk*512 + lane*16 + h] = W[(chunk*32 + (h>=8)*16 + (lane>=16)*8 + h%8)][n0 + lane%16]
  for (int i = threadIdx.x; i < K * 16; i += 128) {
    const int chunk = i >> 9;
    const int rem   = i & 511;
    const int ln    = rem >> 4;
    const int h     = rem & 15;
    const int kk    = chunk * 32 + ((h >> 3) << 4) + ((ln >> 4) << 3) + (h & 7);
    const int nn    = ln & 15;
    lw[i] = (_Float16)W[(size_t)kk * ldw + n0 + nn];
  }
  __syncthreads();

  const int wave = threadIdx.x >> 5;
  const int lane = threadIdx.x & 31;
  const int m0   = (blockIdx.y * 4 + wave) * 16;
  if (m0 >= M) return;                       // wave-uniform (M % 16 == 0)

  const int row = m0 + (lane & 15);
  const int hi  = lane >> 4;
  const int n   = lane & 15;

  v8f acc = {};
#pragma unroll
  for (int kb = 0; kb < K; kb += 32) {
    // A fragment (16-bit A 16x32 layout): two contiguous 8-float runs.
    const float* ap = A + (size_t)row * lda + kb + (hi << 3);
    float4 x0 = *(const float4*)(ap + 0);
    float4 x1 = *(const float4*)(ap + 4);
    float4 x2 = *(const float4*)(ap + 16);
    float4 x3 = *(const float4*)(ap + 20);
    v16h a;
    a[0]=(_Float16)x0.x;  a[1]=(_Float16)x0.y;  a[2]=(_Float16)x0.z;  a[3]=(_Float16)x0.w;
    a[4]=(_Float16)x1.x;  a[5]=(_Float16)x1.y;  a[6]=(_Float16)x1.z;  a[7]=(_Float16)x1.w;
    a[8]=(_Float16)x2.x;  a[9]=(_Float16)x2.y;  a[10]=(_Float16)x2.z; a[11]=(_Float16)x2.w;
    a[12]=(_Float16)x3.x; a[13]=(_Float16)x3.y; a[14]=(_Float16)x3.z; a[15]=(_Float16)x3.w;

    // B fragment: one contiguous 32-byte LDS read.
    const v16h b = *(const v16h*)(lw + (kb << 4) + (lane << 4));

    acc = __builtin_amdgcn_wmma_f32_16x16x32_f16(false, a, false, b,
                                                 (short)0, acc, false, false);
  }

  const float bv = (bias != nullptr) ? bias[n0 + n] : 0.f;
  float* cp = C + (size_t)(m0 + (hi << 3)) * ldc + n0 + n;
#pragma unroll
  for (int r = 0; r < 8; ++r) {
    if (ACC) cp[0] += acc[r];
    else     cp[0] = acc[r] + bv;
    cp += ldc;
  }
}

// ---------------------------------------------------------------------------
// Helpers
// ---------------------------------------------------------------------------
__device__ __forceinline__ void atomicMaxF32(float* addr, float val) {
  if (!(val == val)) return;
  const int iv = __float_as_int(val);
  if (iv >= 0) atomicMax((int*)addr, iv);
  else         atomicMin((unsigned int*)addr, (unsigned)iv);
}

__global__ void fill_f32(float* __restrict__ p, int n, float val) {
  int t = blockIdx.x * blockDim.x + threadIdx.x;
  if (t < n) p[t] = val;
}

__global__ void relu_inplace(float* __restrict__ x, int n) {
  int t = blockIdx.x * blockDim.x + threadIdx.x;
  if (t < n) x[t] = fmaxf(x[t], 0.f);
}

// WeT[h][k][c] = We[c][h*dh + k]   (per-head transposed edge-projection)
__global__ void repack_WeT(const float* __restrict__ We, float* __restrict__ WeT,
                           int H, int dh, int C) {
  int t = blockIdx.x * blockDim.x + threadIdx.x;
  const int tot = H * dh * EDGE_DIM;
  if (t >= tot) return;
  const int h = t / (dh * EDGE_DIM);
  const int r = t % (dh * EDGE_DIM);
  const int k = r / EDGE_DIM;
  const int c = r % EDGE_DIM;
  WeT[t] = We[(size_t)c * C + h * dh + k];
}

// ---------------------------------------------------------------------------
// Edge pass B: alpha = (q[dst,h]·k[src,h] + er·P[dst,h]) / sqrt(dh); seg-max
//   where P[n,h,:] = We_h^T q[n,h,:] was precomputed by WMMA.
// ---------------------------------------------------------------------------
template <int H, int DH>
__global__ __launch_bounds__(256) void edge_alpha(
    const int* __restrict__ src, const int* __restrict__ dst,
    const float* __restrict__ ef, const float* __restrict__ q,
    const float* __restrict__ k, const float* __restrict__ P,
    float* __restrict__ alpha, float* __restrict__ mbuf)
{
  const int t = blockIdx.x * blockDim.x + threadIdx.x;
  if (t >= N_EDGES * H) return;
  const int e = t / H, h = t % H;
  const int sN = src[e], dN = dst[e];
  if (e + 128 < N_EDGES)
    __builtin_prefetch(ef + (size_t)(e + 128) * EDGE_DIM, 0, 1);

  const float* qr = q + (size_t)dN * (H * DH) + h * DH;
  const float* kr = k + (size_t)sN * (H * DH) + h * DH;
  const float* pr = P + (size_t)dN * (H * EDGE_DIM) + h * EDGE_DIM;
  const float* er = ef + (size_t)e * EDGE_DIM;

  float acc = 0.f;
#pragma unroll
  for (int j = 0; j < DH; ++j) acc += qr[j] * kr[j];
#pragma unroll
  for (int c = 0; c < EDGE_DIM; ++c) acc += er[c] * pr[c];

  const float a = acc * (1.0f / sqrtf((float)DH));
  alpha[t] = a;
  atomicMaxF32(mbuf + (size_t)dN * H + h, a);
}

// ---------------------------------------------------------------------------
// Edge pass C: ea = exp(alpha - m[dst]); segment-sum
// ---------------------------------------------------------------------------
template <int H>
__global__ __launch_bounds__(256) void edge_exp(
    const int* __restrict__ dst, const float* __restrict__ mbuf,
    float* __restrict__ alpha, float* __restrict__ sbuf)
{
  const int t = blockIdx.x * blockDim.x + threadIdx.x;
  if (t >= N_EDGES * H) return;
  const int e = t / H, h = t % H;
  const int dN = dst[e];
  const float ea = __expf(alpha[t] - mbuf[(size_t)dN * H + h]);
  alpha[t] = ea;
  atomicAdd(sbuf + (size_t)dN * H + h, ea);
}

// ---------------------------------------------------------------------------
// Edge pass D: w = ea/s;  F[dst,h,:] += w*er ;  out[dst,h,:] += w*v[src,h,:]
//   (the edge-value contribution  Σ w*ek  becomes  F @ We_h  afterwards)
// ---------------------------------------------------------------------------
template <int H, int DH>
__global__ __launch_bounds__(256) void edge_agg(
    const int* __restrict__ src, const int* __restrict__ dst,
    const float* __restrict__ ef, const float* __restrict__ v,
    const float* __restrict__ alpha, const float* __restrict__ sbuf,
    float* __restrict__ F, float* __restrict__ out)
{
  const int t = blockIdx.x * blockDim.x + threadIdx.x;
  if (t >= N_EDGES * H) return;
  const int e = t / H, h = t % H;
  const int sN = src[e], dN = dst[e];
  if (e + 128 < N_EDGES)
    __builtin_prefetch(ef + (size_t)(e + 128) * EDGE_DIM, 0, 1);

  const float w = alpha[t] / (sbuf[(size_t)dN * H + h] + 1e-16f);

  const float* er = ef + (size_t)e * EDGE_DIM;
  float* fr = F + (size_t)dN * (H * EDGE_DIM) + h * EDGE_DIM;
#pragma unroll
  for (int c = 0; c < EDGE_DIM; ++c) atomicAdd(fr + c, w * er[c]);

  const float* vr = v + (size_t)sN * (H * DH) + h * DH;
  float* orow = out + (size_t)dN * (H * DH) + h * DH;
#pragma unroll
  for (int j = 0; j < DH; ++j) atomicAdd(orow + j, w * vr[j]);
}

// ---------------------------------------------------------------------------
extern "C" void kernel_launch(void* const* d_in, const int* in_sizes, int n_in,
                              void* d_out, int out_size, void* d_ws, size_t ws_size,
                              hipStream_t stream) {
  (void)in_sizes; (void)n_in; (void)out_size; (void)ws_size;

  const float* x   = (const float*)d_in[0];
  const int*   ei  = (const int*)d_in[1];
  const float* ef  = (const float*)d_in[2];
  const float* Wq1 = (const float*)d_in[3];  const float* bq1 = (const float*)d_in[4];
  const float* Wk1 = (const float*)d_in[5];  const float* bk1 = (const float*)d_in[6];
  const float* Wv1 = (const float*)d_in[7];  const float* bv1 = (const float*)d_in[8];
  const float* We1 = (const float*)d_in[9];
  const float* Ws1 = (const float*)d_in[10]; const float* bs1 = (const float*)d_in[11];
  const float* Wq2 = (const float*)d_in[12]; const float* bq2 = (const float*)d_in[13];
  const float* Wk2 = (const float*)d_in[14]; const float* bk2 = (const float*)d_in[15];
  const float* Wv2 = (const float*)d_in[16]; const float* bv2 = (const float*)d_in[17];
  const float* We2 = (const float*)d_in[18];
  const float* Ws2 = (const float*)d_in[19]; const float* bs2 = (const float*)d_in[20];

  const int* srcp = ei;
  const int* dstp = ei + N_EDGES;

  // ---------------- workspace carve-out (~223 MB) ----------------
  float* p   = (float*)d_ws;
  float* q1  = p; p += (size_t)N_NODES * C1;
  float* k1  = p; p += (size_t)N_NODES * C1;
  float* v1  = p; p += (size_t)N_NODES * C1;
  float* hb  = p; p += (size_t)N_NODES * C1;        // conv1 out (skip-init)
  float* P1  = p; p += (size_t)N_NODES * (4 * EDGE_DIM);
  float* F1  = p; p += (size_t)N_NODES * (4 * EDGE_DIM);
  float* m1  = p; p += (size_t)N_NODES * 4;
  float* s1  = p; p += (size_t)N_NODES * 4;
  float* a1  = p; p += (size_t)N_EDGES * 4;
  float* q2  = p; p += (size_t)N_NODES * OUT_DIM;
  float* k2  = p; p += (size_t)N_NODES * OUT_DIM;
  float* v2  = p; p += (size_t)N_NODES * OUT_DIM;
  float* P2  = p; p += (size_t)N_NODES * EDGE_DIM;
  float* F2  = p; p += (size_t)N_NODES * EDGE_DIM;
  float* m2  = p; p += (size_t)N_NODES;
  float* s2  = p; p += (size_t)N_NODES;
  float* a2  = p; p += (size_t)N_EDGES;
  float* WeT1 = p; p += (size_t)4 * 32 * EDGE_DIM;  // [H][dh][32]
  float* WeT2 = p; p += (size_t)1 * 64 * EDGE_DIM;

  const int gy = (N_NODES + 63) / 64;
  const dim3 blk(128);

  // ================= layer 1 (H=4, dh=32, C=128) =================
  {
    const dim3 g(C1 / 16, gy, 1);
    gemm_wmma<128, false><<<g, blk, 0, stream>>>(x, IN_DIM, 0, Wq1, C1, 0, bq1, q1, C1, 0, N_NODES);
    gemm_wmma<128, false><<<g, blk, 0, stream>>>(x, IN_DIM, 0, Wk1, C1, 0, bk1, k1, C1, 0, N_NODES);
    gemm_wmma<128, false><<<g, blk, 0, stream>>>(x, IN_DIM, 0, Wv1, C1, 0, bv1, v1, C1, 0, N_NODES);
    gemm_wmma<128, false><<<g, blk, 0, stream>>>(x, IN_DIM, 0, Ws1, C1, 0, bs1, hb, C1, 0, N_NODES);
  }
  repack_WeT<<<(4 * 32 * EDGE_DIM + 255) / 256, 256, 0, stream>>>(We1, WeT1, 4, 32, C1);
  {
    // P1[n, h*32+c] = sum_j q1[n, h*32+j] * WeT1[h][j][c]   (per-head, K=32)
    const dim3 g(EDGE_DIM / 16, gy, 4);
    gemm_wmma<32, false><<<g, blk, 0, stream>>>(q1, C1, 32, WeT1, EDGE_DIM, 32 * EDGE_DIM,
                                                nullptr, P1, 4 * EDGE_DIM, EDGE_DIM, N_NODES);
  }
  fill_f32<<<(N_NODES * 4 + 255) / 256, 256, 0, stream>>>(m1, N_NODES * 4, -INFINITY);
  fill_f32<<<(N_NODES * 4 + 255) / 256, 256, 0, stream>>>(s1, N_NODES * 4, 0.f);
  fill_f32<<<(N_NODES * 128 + 255) / 256, 256, 0, stream>>>(F1, N_NODES * 128, 0.f);
  {
    const int eh = N_EDGES * 4, gb = (eh + 255) / 256;
    edge_alpha<4, 32><<<gb, 256, 0, stream>>>(srcp, dstp, ef, q1, k1, P1, a1, m1);
    edge_exp<4><<<gb, 256, 0, stream>>>(dstp, m1, a1, s1);
    edge_agg<4, 32><<<gb, 256, 0, stream>>>(srcp, dstp, ef, v1, a1, s1, F1, hb);
  }
  {
    // hb[n, h*32+j] += sum_c F1[n, h*32+c] * We1[c][h*32+j]   (per-head, K=32)
    const dim3 g(32 / 16, gy, 4);
    gemm_wmma<32, true><<<g, blk, 0, stream>>>(F1, 4 * EDGE_DIM, EDGE_DIM, We1, C1, 32,
                                               nullptr, hb, C1, 32, N_NODES);
  }
  relu_inplace<<<(N_NODES * C1 + 255) / 256, 256, 0, stream>>>(hb, N_NODES * C1);

  // ================= layer 2 (H=1, dh=64, C=64) =================
  float* outp = (float*)d_out;
  {
    const dim3 g(OUT_DIM / 16, gy, 1);
    gemm_wmma<128, false><<<g, blk, 0, stream>>>(hb, C1, 0, Wq2, OUT_DIM, 0, bq2, q2, OUT_DIM, 0, N_NODES);
    gemm_wmma<128, false><<<g, blk, 0, stream>>>(hb, C1, 0, Wk2, OUT_DIM, 0, bk2, k2, OUT_DIM, 0, N_NODES);
    gemm_wmma<128, false><<<g, blk, 0, stream>>>(hb, C1, 0, Wv2, OUT_DIM, 0, bv2, v2, OUT_DIM, 0, N_NODES);
    gemm_wmma<128, false><<<g, blk, 0, stream>>>(hb, C1, 0, Ws2, OUT_DIM, 0, bs2, outp, OUT_DIM, 0, N_NODES);
  }
  repack_WeT<<<(64 * EDGE_DIM + 255) / 256, 256, 0, stream>>>(We2, WeT2, 1, 64, OUT_DIM);
  {
    // P2[n, c] = sum_j q2[n, j] * WeT2[j][c]   (K=64)
    const dim3 g(EDGE_DIM / 16, gy, 1);
    gemm_wmma<64, false><<<g, blk, 0, stream>>>(q2, OUT_DIM, 0, WeT2, EDGE_DIM, 0,
                                                nullptr, P2, EDGE_DIM, 0, N_NODES);
  }
  fill_f32<<<(N_NODES + 255) / 256, 256, 0, stream>>>(m2, N_NODES, -INFINITY);
  fill_f32<<<(N_NODES + 255) / 256, 256, 0, stream>>>(s2, N_NODES, 0.f);
  fill_f32<<<(N_NODES * EDGE_DIM + 255) / 256, 256, 0, stream>>>(F2, N_NODES * EDGE_DIM, 0.f);
  {
    const int gb = (N_EDGES + 255) / 256;
    edge_alpha<1, 64><<<gb, 256, 0, stream>>>(srcp, dstp, ef, q2, k2, P2, a2, m2);
    edge_exp<1><<<gb, 256, 0, stream>>>(dstp, m2, a2, s2);
    edge_agg<1, 64><<<gb, 256, 0, stream>>>(srcp, dstp, ef, v2, a2, s2, F2, outp);
  }
  {
    // outp[n, j] += sum_c F2[n, c] * We2[c][j]   (K=32)
    const dim3 g(OUT_DIM / 16, gy, 1);
    gemm_wmma<32, true><<<g, blk, 0, stream>>>(F2, EDGE_DIM, 0, We2, OUT_DIM, 0,
                                               nullptr, outp, OUT_DIM, 0, N_NODES);
  }
}
